// RNNAutoencoder_6236292514047
// MI455X (gfx1250) — matrix operand order repaired and checked by reading, not compiled, not run on hardware
//
#include <hip/hip_runtime.h>
#include <hip/hip_bf16.h>

// ---------------------------------------------------------------------------
// GRU autoencoder for MI455X (gfx1250, wave32, WMMA).
//   B=4096, S=512, I=32, H=128.
// Persistent kernel: 256 workgroups x 256 threads; each workgroup owns a
// 16-row batch tile and runs the full 1024-step recurrence.
// Per step: [16x160] x [160x384] GEMM on v_wmma_f32_16x16x32_bf16 (f32 acc),
// weights resident in VGPRs, h staged through ping-pong LDS as bf16.
// This revision: (a) software-pipelined x_t prefetch (raw f32 in registers,
// converted one step later) to hide global-load latency behind the WMMA
// chain, (b) ping-pong LDS h/x buffers so the encoder needs only ONE
// workgroup barrier per step and the decoder two.
// ---------------------------------------------------------------------------

#define HID 128
#define NIN 32
#define SEQ 512
#define BATCH 4096

typedef __attribute__((ext_vector_type(16))) __bf16 v16bf;
typedef __attribute__((ext_vector_type(2)))  __bf16 v2bf;
typedef __attribute__((ext_vector_type(8)))  float  v8f;

union F16 { v16bf v; unsigned int u[8]; };
union P16 { v2bf  v; unsigned int u;    };

// ---- fast scalar f32 -> bf16 (round-to-nearest, cheap) ---------------------
__device__ __forceinline__ unsigned short f2bf(float f) {
    return (unsigned short)((__float_as_uint(f) + 0x8000u) >> 16);
}

// ---- packed f32x2 -> bf16x2 (hardware cvt when available) ------------------
__device__ __forceinline__ unsigned int pack2_bf16(float a, float b) {
#if __has_builtin(__builtin_amdgcn_cvt_pk_bf16_f32)
    P16 p;
    p.v = __builtin_amdgcn_cvt_pk_bf16_f32(a, b);
    return p.u;
#else
    return (unsigned int)f2bf(a) | ((unsigned int)f2bf(b) << 16);
#endif
}

__device__ __forceinline__ v8f splat8(float v) {
    v8f r;
#pragma unroll
    for (int i = 0; i < 8; ++i) r[i] = v;
    return r;
}

__device__ __forceinline__ v8f wmma_bf16(v16bf a, v16bf b, v8f c) {
    return __builtin_amdgcn_wmma_f32_16x16x32_bf16(
        /*neg_a=*/false, a, /*neg_b=*/false, b,
        /*c_mod=*/(short)0, c, /*reuse_a=*/false, /*reuse_b=*/false);
}

// ---- fast activations on the v_exp/v_rcp/v_tanh HW paths -------------------
#define LOG2E 1.4426950408889634f

__device__ __forceinline__ float fast_sigmoid(float x) {
    float e = __builtin_amdgcn_exp2f(-x * LOG2E);
    return __builtin_amdgcn_rcpf(1.0f + e);
}

__device__ __forceinline__ float fast_tanh(float x) {
#if __has_builtin(__builtin_amdgcn_tanhf)
    return __builtin_amdgcn_tanhf(x);        // v_tanh_f32 (CDNA5 transcendental)
#else
    float e = __builtin_amdgcn_exp2f(-2.0f * LOG2E * x);
    return __builtin_fmaf(2.0f, __builtin_amdgcn_rcpf(1.0f + e), -1.0f);
#endif
}

// B-matrix fragment (K x 16, bf16) from a row-major bf16 weight [rows, Kd].
// Lane layout: N = lane&15, VGPR v holds K-pair K = k0 + 2*(v&3) + 8*(lane>>4) + 16*(v>>2).
__device__ __forceinline__ v16bf load_bfrag(const unsigned short* __restrict__ W,
                                            int j, int Kd, int k0, int hi) {
    F16 f;
#pragma unroll
    for (int v = 0; v < 8; ++v) {
        int K = k0 + 2 * (v & 3) + 8 * hi + 16 * (v >> 2);
        f.u[v] = *(const unsigned int*)(W + (size_t)j * Kd + K);
    }
    return f.v;
}

// A-matrix fragment (16 x 32, bf16) from LDS buffer [16 rows, rowstride].
// Lane layout: M = lane&15, VGPR v holds K-pair
//   K = k0 + 2*(v&3) + 8*(lane>>4) + 16*(v>>2).   (vectorizes to ds_load_b128)
__device__ __forceinline__ v16bf load_afrag_lds(const unsigned short* buf,
                                                int row, int rowstride, int k0, int hi) {
    F16 f;
#pragma unroll
    for (int v = 0; v < 8; ++v) {
        int K = k0 + 2 * (v & 3) + 8 * hi + 16 * (v >> 2);
        f.u[v] = *(const unsigned int*)(buf + row * rowstride + K);
    }
    return f.v;
}

// ---- software-pipelined raw x_t tile: 16 floats per lane, two runs of 8 --------
struct XRaw { float4 a0, a1, a2, a3; };

__device__ __forceinline__ XRaw load_x_raw(const float* __restrict__ xrow, int hi) {
    // lane needs floats [8*hi, 8*hi+8) and [16+8*hi, 16+8*hi+8)
    XRaw r;
    const float4* p0 = (const float4*)(xrow + 8 * hi);
    const float4* p1 = (const float4*)(xrow + 16 + 8 * hi);
    r.a0 = p0[0]; r.a1 = p0[1];
    r.a2 = p1[0]; r.a3 = p1[1];
    return r;
}

__device__ __forceinline__ v16bf cvt_x(const XRaw& r) {
    F16 f;
    f.u[0] = pack2_bf16(r.a0.x, r.a0.y);
    f.u[1] = pack2_bf16(r.a0.z, r.a0.w);
    f.u[2] = pack2_bf16(r.a1.x, r.a1.y);
    f.u[3] = pack2_bf16(r.a1.z, r.a1.w);
    f.u[4] = pack2_bf16(r.a2.x, r.a2.y);
    f.u[5] = pack2_bf16(r.a2.z, r.a2.w);
    f.u[6] = pack2_bf16(r.a3.x, r.a3.y);
    f.u[7] = pack2_bf16(r.a3.z, r.a3.w);
    return f.v;
}

// One GRU cell step for this wave's [16 batch x 16 unit] tile.
// 15 WMMAs: r (1 ih + 4 hh), z (1 + 4), i_n (1), h_n (4).
__device__ __forceinline__ v8f gru_compute(const v16bf& ax, const v16bf ah[4],
                                           const v16bf wih[3], const v16bf whh[3][4],
                                           float br, float bz, float bni, float bnh,
                                           const v8f& hprev) {
    v8f aR  = splat8(br);
    v8f aZ  = splat8(bz);
    v8f aNI = splat8(bni);
    v8f aNH = splat8(bnh);
    aR  = wmma_bf16(ax, wih[0], aR);
    aZ  = wmma_bf16(ax, wih[1], aZ);
    aNI = wmma_bf16(ax, wih[2], aNI);
#pragma unroll
    for (int t = 0; t < 4; ++t) {
        aR  = wmma_bf16(ah[t], whh[0][t], aR);
        aZ  = wmma_bf16(ah[t], whh[1][t], aZ);
        aNH = wmma_bf16(ah[t], whh[2][t], aNH);
    }
    v8f h;
#pragma unroll
    for (int i = 0; i < 8; ++i) {
        float r = fast_sigmoid(aR[i]);
        float z = fast_sigmoid(aZ[i]);
        float n = fast_tanh(__builtin_fmaf(r, aNH[i], aNI[i]));
        h[i] = __builtin_fmaf(z, hprev[i] - n, n);   // (1-z)*n + z*h
    }
    return h;
}

__global__ void convert_bf16_kernel(const float* __restrict__ src,
                                    unsigned short* __restrict__ dst, int n) {
    int i = blockIdx.x * blockDim.x + threadIdx.x;
    if (i < n) {
        unsigned int u = __float_as_uint(src[i]);
        u += 0x7FFFu + ((u >> 16) & 1u);             // RNE for weights
        dst[i] = (unsigned short)(u >> 16);
    }
}

__global__ __launch_bounds__(256)
void gru_autoencoder_kernel(
    const float* __restrict__ x, const float* __restrict__ h0,
    const unsigned short* __restrict__ encWih, const unsigned short* __restrict__ encWhh,
    const float* __restrict__ enc_bih, const float* __restrict__ enc_bhh,
    const unsigned short* __restrict__ decWih, const unsigned short* __restrict__ decWhh,
    const float* __restrict__ dec_bih, const float* __restrict__ dec_bhh,
    const unsigned short* __restrict__ linW, const float* __restrict__ lin_b,
    float* __restrict__ out, float* __restrict__ hout)
{
    __shared__ unsigned short hbuf[2][16 * HID];  // ping-pong h tile, bf16
    __shared__ unsigned short xbuf[2][16 * NIN];  // ping-pong decoder input

    const int tid  = threadIdx.x;
    const int lane = tid & 31;
    const int wave = tid >> 5;          // 0..7, owns units [wave*16, wave*16+16)
    const int hi   = lane >> 4;
    const int M    = lane & 15;         // A-frag row / C-layout N
    const int b0   = blockIdx.x * 16;   // batch tile base
    const int u    = wave * 16 + M;     // this lane's hidden-unit column

    // ---------------- encoder weights -> VGPRs (persistent) ----------------
    v16bf wih[3], whh[3][4];
#pragma unroll
    for (int g = 0; g < 3; ++g) {
        wih[g] = load_bfrag(encWih, g * HID + u, NIN, 0, hi);
#pragma unroll
        for (int t = 0; t < 4; ++t)
            whh[g][t] = load_bfrag(encWhh, g * HID + u, HID, 32 * t, hi);
    }
    float br  = enc_bih[u]           + enc_bhh[u];
    float bz  = enc_bih[HID + u]     + enc_bhh[HID + u];
    float bni = enc_bih[2 * HID + u];
    float bnh = enc_bhh[2 * HID + u];

    // ---------------- h0 -> registers (C layout) + LDS ---------------------
    v8f hprev;
#pragma unroll
    for (int r = 0; r < 8; ++r) {
        int row = r + 8 * hi;
        hprev[r] = h0[(size_t)(b0 + row) * HID + u];
        hbuf[0][row * HID + u] = f2bf(hprev[r]);
    }
    __syncthreads();

    const float* xrow = x + (size_t)(b0 + M) * SEQ * NIN;  // per-lane batch row

    // ---------------- encoder scan (1 barrier/step, x prefetched) ----------
    int p = 0;
    XRaw xr = load_x_raw(xrow, hi);                 // s = 0
    for (int s = 0; s < SEQ; ++s) {
        int sn = (s + 1 < SEQ) ? s + 1 : s;
        XRaw xnext = load_x_raw(xrow + (size_t)sn * NIN, hi);   // prefetch s+1

        v16bf ax = cvt_x(xr);
        v16bf ah[4];
#pragma unroll
        for (int t = 0; t < 4; ++t) ah[t] = load_afrag_lds(hbuf[p], M, HID, 32 * t, hi);
        hprev = gru_compute(ax, ah, wih, whh, br, bz, bni, bnh, hprev);
#pragma unroll
        for (int r = 0; r < 8; ++r)
            hbuf[p ^ 1][(r + 8 * hi) * HID + u] = f2bf(hprev[r]);
        __syncthreads();                 // new h visible; old reads done
        p ^= 1;
        xr = xnext;
    }

    // encoder final hidden state -> second output
#pragma unroll
    for (int r = 0; r < 8; ++r)
        hout[(size_t)(b0 + r + 8 * hi) * HID + u] = hprev[r];

    // ---------------- decoder weights -> VGPRs (reuse registers) -----------
#pragma unroll
    for (int g = 0; g < 3; ++g) {
        wih[g] = load_bfrag(decWih, g * HID + u, NIN, 0, hi);
#pragma unroll
        for (int t = 0; t < 4; ++t)
            whh[g][t] = load_bfrag(decWhh, g * HID + u, HID, 32 * t, hi);
    }
    br  = dec_bih[u]           + dec_bhh[u];
    bz  = dec_bih[HID + u]     + dec_bhh[HID + u];
    bni = dec_bih[2 * HID + u];
    bnh = dec_bhh[2 * HID + u];

    // output projection weights for waves 0,1 (16 output cols each)
    v16bf wlin[4];
    float blin = 0.0f;
    const int jy = (wave & 1) * 16 + M;  // valid when wave < 2
    if (wave < 2) {
#pragma unroll
        for (int t = 0; t < 4; ++t) wlin[t] = load_bfrag(linW, jy, HID, 32 * t, hi);
        blin = lin_b[jy];
    }

    // seed: hbuf[0] := final encoder h ; xbuf[0] := x[:, 0, :]
#pragma unroll
    for (int r = 0; r < 8; ++r)
        hbuf[0][(r + 8 * hi) * HID + u] = f2bf(hprev[r]);
    for (int e = tid; e < 16 * NIN; e += 256) {
        int row = e >> 5, col = e & 31;
        xbuf[0][e] = f2bf(x[(size_t)(b0 + row) * SEQ * NIN + col]);
    }
    __syncthreads();

    // ---------------- decoder scan (autoregressive, 2 barriers/step) -------
    p = 0;
    for (int s = 0; s < SEQ; ++s) {
        v16bf ax = load_afrag_lds(xbuf[p], M, NIN, 0, hi);
        v16bf ah[4];
#pragma unroll
        for (int t = 0; t < 4; ++t) ah[t] = load_afrag_lds(hbuf[p], M, HID, 32 * t, hi);
        hprev = gru_compute(ax, ah, wih, whh, br, bz, bni, bnh, hprev);
#pragma unroll
        for (int r = 0; r < 8; ++r)
            hbuf[p ^ 1][(r + 8 * hi) * HID + u] = f2bf(hprev[r]);
        __syncthreads();                 // h_new complete

        if (wave < 2) {                  // y = h_new @ linW^T + lin_b
            v8f aY = splat8(blin);
#pragma unroll
            for (int t = 0; t < 4; ++t) {
                v16bf ahn = load_afrag_lds(hbuf[p ^ 1], M, HID, 32 * t, hi);
                aY = wmma_bf16(ahn, wlin[t], aY);
            }
#pragma unroll
            for (int r = 0; r < 8; ++r) {
                int row = r + 8 * hi;
                out[(size_t)(b0 + row) * SEQ * NIN + (size_t)s * NIN + jy] = aY[r];
                xbuf[p ^ 1][row * NIN + jy] = f2bf(aY[r]);   // next x
            }
        }
        __syncthreads();                 // y/xbuf visible for next step
        p ^= 1;
    }
}

extern "C" void kernel_launch(void* const* d_in, const int* in_sizes, int n_in,
                              void* d_out, int out_size, void* d_ws, size_t ws_size,
                              hipStream_t stream) {
    (void)in_sizes; (void)n_in; (void)out_size; (void)ws_size;

    const float* x        = (const float*)d_in[0];
    const float* h0       = (const float*)d_in[1];
    const float* enc_Wih  = (const float*)d_in[2];
    const float* enc_Whh  = (const float*)d_in[3];
    const float* enc_bih  = (const float*)d_in[4];
    const float* enc_bhh  = (const float*)d_in[5];
    const float* dec_Wih  = (const float*)d_in[6];
    const float* dec_Whh  = (const float*)d_in[7];
    const float* dec_bih  = (const float*)d_in[8];
    const float* dec_bhh  = (const float*)d_in[9];
    const float* lin_W    = (const float*)d_in[10];
    const float* lin_b    = (const float*)d_in[11];

    // bf16 weight staging in workspace
    unsigned short* ws   = (unsigned short*)d_ws;
    unsigned short* eWih = ws;                       // 384*32
    unsigned short* eWhh = eWih + 3 * HID * NIN;     // 384*128
    unsigned short* dWih = eWhh + 3 * HID * HID;
    unsigned short* dWhh = dWih + 3 * HID * NIN;
    unsigned short* wLin = dWhh + 3 * HID * HID;     // 32*128

    auto conv = [&](const float* src, unsigned short* dst, int n) {
        convert_bf16_kernel<<<(n + 255) / 256, 256, 0, stream>>>(src, dst, n);
    };
    conv(enc_Wih, eWih, 3 * HID * NIN);
    conv(enc_Whh, eWhh, 3 * HID * HID);
    conv(dec_Wih, dWih, 3 * HID * NIN);
    conv(dec_Whh, dWhh, 3 * HID * HID);
    conv(lin_W,  wLin, NIN * HID);

    float* out  = (float*)d_out;                          // [B, S, I]
    float* hout = out + (size_t)BATCH * SEQ * NIN;        // [B, H]

    gru_autoencoder_kernel<<<BATCH / 16, 256, 0, stream>>>(
        x, h0, eWih, eWhh, enc_bih, enc_bhh,
        dWih, dWhh, dec_bih, dec_bhh, wLin, lin_b, out, hout);
}
